// EncoderRNNWithVector_59511066853855
// MI455X (gfx1250) — matrix operand order repaired and checked by reading, compiled, not run on hardware
//
#include <hip/hip_runtime.h>

// ---------------------------------------------------------------------------
// Problem dims (fixed by the reference): B=64, T=256, D_in=H=D_out=1024.
// ---------------------------------------------------------------------------
#define B_DIM 64
#define T_DIM 256
#define H_DIM 1024
#define G_DIM 3072   // 3*H

typedef __bf16 v16bf __attribute__((ext_vector_type(16)));
typedef __bf16 v8bf  __attribute__((ext_vector_type(8)));
typedef float  v8f   __attribute__((ext_vector_type(8)));

// ---------------------------------------------------------------------------
// WMMA fragment loaders (wave32, 16x16x32 bf16).
// A (MxK row-major): lane l -> row = l&15; two contiguous 8-elem K runs at
//   k0 + 16*(l>>4) and k0 + 16*(l>>4) + 8    (matches B's contiguous K order)
// B supplied as (N,K) row-major (i.e. W itself): lane l -> col n = l&15,
//   contiguous 16 bf16 at k0 + 16*(l>>4)      (same contiguous K order as A)
// ---------------------------------------------------------------------------
__device__ __forceinline__ v16bf load_frag_a(const __bf16* __restrict__ A,
                                             int lda, int row0, int k0, int lane) {
  int r = row0 + (lane & 15);
  int h = lane >> 4;
  const __bf16* base = A + (size_t)r * lda + k0;
  v8bf lo = *(const v8bf*)(base + 16 * h);
  v8bf hi = *(const v8bf*)(base + 16 * h + 8);
  return __builtin_shufflevector(lo, hi, 0,1,2,3,4,5,6,7,8,9,10,11,12,13,14,15);
}

__device__ __forceinline__ v16bf load_frag_b(const __bf16* __restrict__ Bt,
                                             int ldb, int n0, int k0, int lane) {
  int n = n0 + (lane & 15);
  int h = lane >> 4;
  return *(const v16bf*)(Bt + (size_t)n * ldb + k0 + 16 * h);
}

__device__ __forceinline__ v8f wmma_bf16(v16bf a, v16bf b, v8f c) {
  return __builtin_amdgcn_wmma_f32_16x16x32_bf16(false, a, false, b, (short)0, c,
                                                 false, false);
}

// ---------------------------------------------------------------------------
// Generic bf16 WMMA GEMM: C[M,N] = A[M,K] * Bt[N,K]^T (+ bias[N]).
// Block tile 64x64, 256 threads = 8 waves, 2 accumulators per wave.
// ---------------------------------------------------------------------------
template <bool OUT_BF16, bool ADD_BIAS>
__global__ __launch_bounds__(256) void gemm_wmma(
    const __bf16* __restrict__ A, const __bf16* __restrict__ Bt,
    void* __restrict__ Cout, const float* __restrict__ bias,
    int M, int N, int K) {
  const int lane = threadIdx.x & 31;
  const int w    = threadIdx.x >> 5;
  const int bm   = blockIdx.y * 64;
  const int bn   = blockIdx.x * 64;
  const int mi   = w & 3;           // 4 M-tiles
  const int nj   = (w >> 2) * 2;    // waves 0-3: N-tiles {0,1}; 4-7: {2,3}
  const int row0 = bm + mi * 16;

  v8f acc0 = {}, acc1 = {};
#pragma unroll 1
  for (int k0 = 0; k0 < K; k0 += 32) {
    v16bf a  = load_frag_a(A, K, row0, k0, lane);
    v16bf b0 = load_frag_b(Bt, K, bn + nj * 16, k0, lane);
    v16bf b1 = load_frag_b(Bt, K, bn + (nj + 1) * 16, k0, lane);
    acc0 = wmma_bf16(a, b0, acc0);
    acc1 = wmma_bf16(a, b1, acc1);
  }

  // C/D layout: VGPR v holds M = row0 + v + 8*(lane>>4), N = lane&15.
  const int rb  = row0 + 8 * (lane >> 4);
  const int nlo = lane & 15;
#pragma unroll
  for (int tile = 0; tile < 2; ++tile) {
    v8f acc = tile ? acc1 : acc0;
    int c = bn + (nj + tile) * 16 + nlo;
    float bv = ADD_BIAS ? bias[c] : 0.0f;
#pragma unroll
    for (int v = 0; v < 8; ++v) {
      float val = acc[v] + bv;
      size_t idx = (size_t)(rb + v) * N + c;
      if (OUT_BF16) ((__bf16*)Cout)[idx] = (__bf16)val;
      else          ((float*)Cout)[idx]  = val;
    }
  }
}

// ---------------------------------------------------------------------------
// Elementwise helpers
// ---------------------------------------------------------------------------
__global__ void cvt_f32_to_bf16(const float* __restrict__ in,
                                __bf16* __restrict__ out, int n) {
  int i = blockIdx.x * blockDim.x + threadIdx.x;
  if (i < n) out[i] = (__bf16)in[i];
}

// out[d*R + h] = in[h*C + d]   (W_in (H,D) -> W_in^T (D,H) in bf16)
__global__ void transpose_f32_to_bf16(const float* __restrict__ in,
                                      __bf16* __restrict__ out, int R, int C) {
  int i = blockIdx.x * blockDim.x + threadIdx.x;
  if (i < R * C) {
    int h = i & (R - 1);
    int d = i >> 10;              // R == C == 1024
    out[(size_t)d * R + h] = (__bf16)in[(size_t)h * C + d];
  }
}

// word_inputs (B,T,D) fp32 -> (T*B, D) bf16
__global__ void permute_bt_to_tb(const float* __restrict__ in,
                                 __bf16* __restrict__ out) {
  int i = blockIdx.x * blockDim.x + threadIdx.x;   // output-linear
  int d  = i & (H_DIM - 1);
  int bt = i >> 10;
  int b  = bt & (B_DIM - 1);
  int t  = bt >> 6;
  out[i] = (__bf16)in[((size_t)b * T_DIM + t) * H_DIM + d];
}

// bc[g] = b_ih[g] + sum_h W_ih[g,h] * b_in[h]   (tiny, one thread per g)
__global__ void fused_gate_bias(const float* __restrict__ W_ih,
                                const float* __restrict__ b_in,
                                const float* __restrict__ b_ih,
                                float* __restrict__ bc) {
  int g = blockIdx.x * blockDim.x + threadIdx.x;
  if (g < G_DIM) {
    float s = b_ih[g];
    const float* row = W_ih + (size_t)g * H_DIM;
    for (int h = 0; h < H_DIM; ++h) s += row[h] * b_in[h];
    bc[g] = s;
  }
}

__global__ void init_h(const float* __restrict__ h0, float* __restrict__ hf,
                       __bf16* __restrict__ hb) {
  int i = blockIdx.x * blockDim.x + threadIdx.x;
  if (i < B_DIM * H_DIM) {
    float v = h0[i];
    hf[i] = v;
    hb[i] = (__bf16)v;
  }
}

__global__ void copy_f32(const float* __restrict__ in, float* __restrict__ out,
                         int n) {
  int i = blockIdx.x * blockDim.x + threadIdx.x;
  if (i < n) out[i] = in[i];
}

__global__ void zero_barrier(unsigned* ctr, unsigned* gen) {
  *ctr = 0u;
  *gen = 0u;
}

// ---------------------------------------------------------------------------
// Device-wide barrier for the persistent recurrence kernel (16 blocks).
// ---------------------------------------------------------------------------
__device__ __forceinline__ void grid_barrier(unsigned* ctr, unsigned* gen,
                                             unsigned nb) {
  __syncthreads();
  if (threadIdx.x == 0) {
    __threadfence();
    unsigned my_gen = atomicAdd(gen, 0u);
    unsigned arrived = atomicAdd(ctr, 1u);
    if (arrived == nb - 1) {
      atomicExch(ctr, 0u);
      __threadfence();
      atomicAdd(gen, 1u);
    } else {
      while (atomicAdd(gen, 0u) == my_gen) { __builtin_amdgcn_s_sleep(1); }
    }
  }
  __syncthreads();
}

__device__ __forceinline__ float sigmoidf_(float x) {
  return 1.0f / (1.0f + __expf(-x));
}

// ---------------------------------------------------------------------------
// Persistent GRU recurrence. 16 WGs x 256 threads; WG j owns h-columns
// [64j, 64j+64) and computes all three gate slices (192 W_hh rows) for them.
// Per step: 4x12 WMMA 16x16 tiles over K=1024, gate fusion in epilogue,
// ping-pong h buffers + one grid barrier per step.
//
// W_hh (6 MB bf16) is L2-resident; it is deliberately re-streamed from L2
// every step. The asm memory clobber at the top of the t-loop stops LLVM from
// hoisting the (t-invariant) W_hh fragment loads out of the loop, which in a
// previous build spilled 6 KB/lane to scratch and serialized every WMMA
// behind an s_wait_loadcnt 0.
// ---------------------------------------------------------------------------
#define HG_STRIDE 196   // 192 + 4 padding -> conflict-free LDS scatter

__global__ __launch_bounds__(256) void gru_recurrence(
    const __bf16* __restrict__ Whh,      // (3H, H) bf16, row-major (N,K)
    const float* __restrict__ XG,        // (T, B, 3H) fp32
    const float* __restrict__ bhh,       // (3H)
    float* __restrict__ hf0, float* __restrict__ hf1,
    __bf16* __restrict__ hb0, __bf16* __restrict__ hb1,
    unsigned* __restrict__ bar_ctr, unsigned* __restrict__ bar_gen) {
  __shared__ float hg[B_DIM * HG_STRIDE];   // 64 x 192 (padded) fp32 ~ 50 KB

  const int j     = blockIdx.x;            // h-column slice owner
  const int lane  = threadIdx.x & 31;
  const int w     = threadIdx.x >> 5;
  const int mi    = w & 3;                 // 4 M-tiles (batch rows)
  const int nbase = (w >> 2) * 6;          // 12 N-tiles split over 2 wave-groups
  const int row0  = mi * 16;

  // Precompute the W_hh row bases for this wave's 6 N-tiles.
  int wrow[6];
#pragma unroll
  for (int q = 0; q < 6; ++q) {
    int ni  = nbase + q;
    wrow[q] = (ni >> 2) * H_DIM + j * 64 + (ni & 3) * 16;
  }

#pragma unroll 1
  for (int t = 0; t < T_DIM; ++t) {
    // Optimization barrier: forces W_hh fragments to be streamed from L2
    // each step instead of being hoisted + spilled to scratch.
    asm volatile("" ::: "memory");

    const float*  hf_cur = (t & 1) ? hf1 : hf0;
    float*        hf_nxt = (t & 1) ? hf0 : hf1;
    const __bf16* hb_cur = (t & 1) ? hb1 : hb0;
    __bf16*       hb_nxt = (t & 1) ? hb0 : hb1;

    // Prefetch next step's x_gates slice (streams from HBM; 201 MB total so
    // it is not L2-resident). 192 x 256B segments per WG, one per thread.
    if (t + 1 < T_DIM) {
      int e = threadIdx.x;
      if (e < B_DIM * 3) {
        int b = e >> 2;                       // 0..63 (e/3 would div; use /3)
        b = e / 3;
        int gate = e - b * 3;
        const float* p = XG + (size_t)(t + 1) * (B_DIM * G_DIM) +
                         (size_t)b * G_DIM + gate * H_DIM + j * 64;
        __builtin_prefetch(p, 0, 1);
      }
    }

    // --- hg = h @ Whh_slice^T  (64 x 192 per WG) -------------------------
    v8f acc[6] = {};
#pragma unroll 1
    for (int k0 = 0; k0 < H_DIM; k0 += 32) {
      // Issue all fragment loads for this K-chunk first (14 x b128 in a
      // clause), then the 6 WMMAs; lets loads overlap with WMMA issue.
      v16bf a = load_frag_a(hb_cur, H_DIM, row0, k0, lane);
      v16bf bfrag[6];
#pragma unroll
      for (int q = 0; q < 6; ++q)
        bfrag[q] = load_frag_b(Whh, H_DIM, wrow[q], k0, lane);
#pragma unroll
      for (int q = 0; q < 6; ++q) acc[q] = wmma_bf16(a, bfrag[q], acc[q]);
    }

    // --- spill accumulators to LDS (C layout scatter) --------------------
    {
      int rb  = row0 + 8 * (lane >> 4);
      int nlo = lane & 15;
#pragma unroll
      for (int q = 0; q < 6; ++q) {
        int c0 = (nbase + q) * 16 + nlo;
#pragma unroll
        for (int v = 0; v < 8; ++v) hg[(rb + v) * HG_STRIDE + c0] = acc[q][v];
      }
    }
    __syncthreads();

    // --- fused gate epilogue ---------------------------------------------
    const float* xg = XG + (size_t)t * (B_DIM * G_DIM);
#pragma unroll 1
    for (int e = threadIdx.x; e < B_DIM * 64; e += 256) {
      int b = e & (B_DIM - 1);
      int c = e >> 6;
      int hc = j * 64 + c;
      float hr = hg[b * HG_STRIDE + c]        + bhh[hc];
      float hz = hg[b * HG_STRIDE + 64 + c]   + bhh[H_DIM + hc];
      float hn = hg[b * HG_STRIDE + 128 + c]  + bhh[2 * H_DIM + hc];
      const float* xrow = xg + (size_t)b * G_DIM;
      float r = sigmoidf_(xrow[hc] + hr);
      float z = sigmoidf_(xrow[H_DIM + hc] + hz);
      float n = tanhf(xrow[2 * H_DIM + hc] + r * hn);
      float hold = hf_cur[b * H_DIM + hc];
      float hnew = (1.0f - z) * n + z * hold;
      hf_nxt[b * H_DIM + hc] = hnew;
      hb_nxt[b * H_DIM + hc] = (__bf16)hnew;
    }

    grid_barrier(bar_ctr, bar_gen, gridDim.x);
  }
}

// ---------------------------------------------------------------------------
// Host-side orchestration
// ---------------------------------------------------------------------------
extern "C" void kernel_launch(void* const* d_in, const int* in_sizes, int n_in,
                              void* d_out, int out_size, void* d_ws,
                              size_t ws_size, hipStream_t stream) {
  (void)in_sizes; (void)n_in; (void)out_size; (void)ws_size;

  const float* word_inputs = (const float*)d_in[0];
  const float* hidden      = (const float*)d_in[1];
  const float* W_in        = (const float*)d_in[2];
  const float* b_in        = (const float*)d_in[3];
  const float* W_ih        = (const float*)d_in[4];
  const float* b_ih        = (const float*)d_in[5];
  const float* W_hh        = (const float*)d_in[6];
  const float* b_hh        = (const float*)d_in[7];
  const float* W_out       = (const float*)d_in[8];
  const float* b_out       = (const float*)d_in[9];
  float* out = (float*)d_out;

  char* ws = (char*)d_ws;
  size_t off = 0;
  auto alloc = [&](size_t bytes) -> void* {
    void* p = ws + off;
    off = (off + bytes + 255) & ~(size_t)255;
    return p;
  };

  __bf16* w_ih_b   = (__bf16*)alloc((size_t)G_DIM * H_DIM * 2);
  __bf16* w_inT_b  = (__bf16*)alloc((size_t)H_DIM * H_DIM * 2);
  __bf16* w_comb_b = (__bf16*)alloc((size_t)G_DIM * H_DIM * 2);
  __bf16* w_hh_b   = (__bf16*)alloc((size_t)G_DIM * H_DIM * 2);
  __bf16* w_out_b  = (__bf16*)alloc((size_t)H_DIM * H_DIM * 2);
  float*  bc       = (float*)alloc((size_t)G_DIM * 4);
  __bf16* xin_b    = (__bf16*)alloc((size_t)T_DIM * B_DIM * H_DIM * 2);
  float*  XG       = (float*)alloc((size_t)T_DIM * B_DIM * G_DIM * 4);
  float*  hf0      = (float*)alloc((size_t)B_DIM * H_DIM * 4);
  float*  hf1      = (float*)alloc((size_t)B_DIM * H_DIM * 4);
  __bf16* hb0      = (__bf16*)alloc((size_t)B_DIM * H_DIM * 2);
  __bf16* hb1      = (__bf16*)alloc((size_t)B_DIM * H_DIM * 2);
  unsigned* bar    = (unsigned*)alloc(256);

  // 1) precision conversion / layout prep
  cvt_f32_to_bf16<<<(G_DIM * H_DIM) / 256, 256, 0, stream>>>(W_ih, w_ih_b, G_DIM * H_DIM);
  transpose_f32_to_bf16<<<(H_DIM * H_DIM) / 256, 256, 0, stream>>>(W_in, w_inT_b, H_DIM, H_DIM);
  cvt_f32_to_bf16<<<(G_DIM * H_DIM) / 256, 256, 0, stream>>>(W_hh, w_hh_b, G_DIM * H_DIM);
  cvt_f32_to_bf16<<<(H_DIM * H_DIM) / 256, 256, 0, stream>>>(W_out, w_out_b, H_DIM * H_DIM);
  permute_bt_to_tb<<<(T_DIM * B_DIM * H_DIM) / 256, 256, 0, stream>>>(word_inputs, xin_b);
  fused_gate_bias<<<(G_DIM + 255) / 256, 256, 0, stream>>>(W_ih, b_in, b_ih, bc);

  // 2) W_comb = W_ih @ W_in   (M=3072, N=1024, K=1024) -> bf16
  gemm_wmma<true, false><<<dim3(H_DIM / 64, G_DIM / 64), 256, 0, stream>>>(
      w_ih_b, w_inT_b, w_comb_b, nullptr, G_DIM, H_DIM, H_DIM);

  // 3) XG = X @ W_comb^T + bc   (M=T*B=16384, N=3072, K=1024) -> fp32 (T,B,3H)
  gemm_wmma<false, true><<<dim3(G_DIM / 64, (T_DIM * B_DIM) / 64), 256, 0, stream>>>(
      xin_b, w_comb_b, XG, bc, T_DIM * B_DIM, G_DIM, H_DIM);

  // 4) recurrence (persistent, 16 WGs, 256 steps)
  init_h<<<(B_DIM * H_DIM) / 256, 256, 0, stream>>>(hidden, hf0, hb0);
  zero_barrier<<<1, 1, 0, stream>>>(bar, bar + 1);
  gru_recurrence<<<H_DIM / 64, 256, 0, stream>>>(w_hh_b, XG, b_hh, hf0, hf1, hb0,
                                                 hb1, bar, bar + 1);

  // T=256 is even -> final state landed in buffer 0.
  // 5) out_last = h_T @ W_out^T + b_out   (M=64, N=1024, K=1024)
  gemm_wmma<false, true><<<dim3(H_DIM / 64, B_DIM / 64), 256, 0, stream>>>(
      hb0, w_out_b, out, b_out, B_DIM, H_DIM, H_DIM);

  // 6) hidden_out = h_T (fp32, exact recurrence state)
  copy_f32<<<(B_DIM * H_DIM) / 256, 256, 0, stream>>>(hf0, out + B_DIM * H_DIM,
                                                      B_DIM * H_DIM);
}
